// RWKV_TimeMix_For_Quant_6536940224802
// MI455X (gfx1250) — matrix operand order repaired and checked by reading, not compiled, not run on hardware
//
#include <hip/hip_runtime.h>
#include <hip/hip_bf16.h>

// ---------------------------------------------------------------------------
// RWKV TimeMix single-step, MI455X (gfx1250, wave32, WMMA + TDM).
// bf16 WMMA (16x16x32, f32 accum) for all 4 GEMMs; WKV state update fused
// into the triple-GEMM epilogue; tile movement via Tensor Data Mover
// (tensor_load_to_lds, TENSORcnt-tracked, double-buffered LDS).
// ---------------------------------------------------------------------------

#define E_DIM 2048
#define B_DIM 8192

constexpr int BM = 128;          // block tile M
constexpr int BN = 128;          // block tile N
constexpr int BK = 32;           // K step per WMMA
constexpr int LDSS = BK + 8;     // padded LDS row stride (bf16) = 80B rows, conflict-free b128
constexpr int NK = E_DIM / BK;
constexpr int TILE_E = BM * LDSS;            // elements per LDS tile (5120)

typedef __attribute__((ext_vector_type(16))) __bf16 v16bf;
typedef __attribute__((ext_vector_type(8)))  float  v8f;
typedef __attribute__((ext_vector_type(4)))  unsigned int u32x4;
typedef __attribute__((ext_vector_type(8)))  int i32x8;
typedef __attribute__((ext_vector_type(4)))  int i32x4;

union FragU { v16bf v; uint4 q[2]; };

__device__ __forceinline__ unsigned short f2bf(float f) {
  unsigned int u = __float_as_uint(f);
  u += 0x7fffu + ((u >> 16) & 1u);            // round-to-nearest-even
  return (unsigned short)(u >> 16);
}

__device__ __forceinline__ uint2 pack_bf4(float a, float b, float c, float d) {
  uint2 r;
  r.x = (unsigned)f2bf(a) | ((unsigned)f2bf(b) << 16);
  r.y = (unsigned)f2bf(c) | ((unsigned)f2bf(d) << 16);
  return r;
}

// ---- TDM: one descriptor moves a 128x32 bf16 tile (row stride E) into LDS,
// inserting 16B padding after every 64B row (-> LDSS=40 bf16 LDS stride).
// D# fields per CDNA5 ISA 8.3/8.4:
//  group0: count=1 | lds_addr | global_addr[56:0] | type=2
//  group1: data_size=2B(code1), pad_enable, pad_interval=16dw(code3),
//          pad_amount=4dw(code3), tensor_dim0=2048, tensor_dim1=8192,
//          tile_dim0=32, tile_dim1=128, tensor_dim0_stride=2048
__device__ __forceinline__ void tdm_load_tile(const void* gaddr, unsigned lds_addr) {
  unsigned long long ga = (unsigned long long)(uintptr_t)gaddr;
  u32x4 g0 = { 1u,                                   // count=1 (valid descriptor)
               lds_addr,                             // LDS byte address
               (unsigned)ga,                         // global addr [31:0]
               (unsigned)((ga >> 32) & 0x01ffffffull) | 0x80000000u }; // [56:32] | type=2
  i32x8 g1 = { (int)0x06D10000,                      // data_size=2B, pad_en, int=16dw, amt=4dw
               (int)(2048u << 16),                   // tensor_dim0[15:0]=2048
               (int)(8192u << 16),                   // tensor_dim0[31:16]=0, tensor_dim1[15:0]=8192
               (int)(32u << 16),                     // tensor_dim1[31:16]=0, tile_dim0=32
               128,                                  // tile_dim1=128, tile_dim2=0
               2048,                                 // tensor_dim0_stride[31:0]
               0, 0 };                               // stride hi, tensor_dim1_stride
  i32x4 gz = { 0, 0, 0, 0 };                         // groups 2/3: 2-D tile, unused
  i32x8 gx = { 0, 0, 0, 0, 0, 0, 0, 0 };             // extended group (unused, zero)
  __builtin_amdgcn_tensor_load_to_lds(g0, g1, gz, gz, gx, 0);
}

// A fragment 16x32 bf16 (M x K): lanes 0-15 hold M=lane, K runs {0..7,16..23};
// lanes 16-31 hold M=lane-16, K runs {8..15,24..31}.  (ISA 7.12.2)
__device__ __forceinline__ v16bf frag_a(const unsigned short* t, int rowBase, int lane) {
  int row = rowBase + (lane & 15);
  int klo = (lane < 16) ? 0 : 8;
  const unsigned short* p = t + row * LDSS + klo;
  FragU f;
  f.q[0] = *(const uint4*)(p);
  f.q[1] = *(const uint4*)(p + 16);
  return f.v;
}

// B fragment 32x16 bf16 (K x N): lane = N, lanes 0-15 hold K runs {0..7,16..23},
// lanes 16-31 hold K runs {8..15,24..31}, 2 packed per VGPR in K order.
__device__ __forceinline__ v16bf frag_b(const unsigned short* t, int rowBase, int lane) {
  int row = rowBase + (lane & 15);
  int klo = (lane < 16) ? 0 : 8;
  const unsigned short* p = t + row * LDSS + klo;
  FragU f;
  f.q[0] = *(const uint4*)(p);
  f.q[1] = *(const uint4*)(p + 16);
  return f.v;
}

// --------------------------- elementwise prep ------------------------------
__global__ void prep_kernel(const float* __restrict__ x,  const float* __restrict__ sx,
                            const float* __restrict__ tmk, const float* __restrict__ tmv,
                            const float* __restrict__ tmr,
                            unsigned short* __restrict__ xk, unsigned short* __restrict__ xv,
                            unsigned short* __restrict__ xr, float* __restrict__ nx) {
  size_t i4 = ((size_t)blockIdx.x * 256 + threadIdx.x) * 4;
  int e = (int)(i4 & (E_DIM - 1));
  float4 xf  = *(const float4*)(x + i4);
  float4 sxf = *(const float4*)(sx + i4);
  float4 mk  = *(const float4*)(tmk + e);
  float4 mv  = *(const float4*)(tmv + e);
  float4 mr  = *(const float4*)(tmr + e);
  *(uint2*)(xk + i4) = pack_bf4(fmaf(mk.x, xf.x - sxf.x, sxf.x), fmaf(mk.y, xf.y - sxf.y, sxf.y),
                                fmaf(mk.z, xf.z - sxf.z, sxf.z), fmaf(mk.w, xf.w - sxf.w, sxf.w));
  *(uint2*)(xv + i4) = pack_bf4(fmaf(mv.x, xf.x - sxf.x, sxf.x), fmaf(mv.y, xf.y - sxf.y, sxf.y),
                                fmaf(mv.z, xf.z - sxf.z, sxf.z), fmaf(mv.w, xf.w - sxf.w, sxf.w));
  *(uint2*)(xr + i4) = pack_bf4(fmaf(mr.x, xf.x - sxf.x, sxf.x), fmaf(mr.y, xf.y - sxf.y, sxf.y),
                                fmaf(mr.z, xf.z - sxf.z, sxf.z), fmaf(mr.w, xf.w - sxf.w, sxf.w));
  *(float4*)(nx + i4) = xf;
}

// ------------------------- weight fp32 -> bf16 -----------------------------
__global__ void wconv_kernel(const float* __restrict__ s0, const float* __restrict__ s1,
                             const float* __restrict__ s2, const float* __restrict__ s3,
                             unsigned short* __restrict__ d0, unsigned short* __restrict__ d1,
                             unsigned short* __restrict__ d2, unsigned short* __restrict__ d3) {
  const float* s; unsigned short* d;
  switch (blockIdx.y) {
    case 0:  s = s0; d = d0; break;
    case 1:  s = s1; d = d1; break;
    case 2:  s = s2; d = d2; break;
    default: s = s3; d = d3; break;
  }
  size_t i4 = ((size_t)blockIdx.x * 256 + threadIdx.x) * 4;
  float4 f = *(const float4*)(s + i4);
  *(uint2*)(d + i4) = pack_bf4(f.x, f.y, f.z, f.w);
}

// ------------- triple NT-GEMM (k,v,rlin) + fused WKV epilogue --------------
__global__ __launch_bounds__(256, 1)
void rwkv_fused_gemm3(const unsigned short* __restrict__ xk,
                      const unsigned short* __restrict__ xv,
                      const unsigned short* __restrict__ xr,
                      const unsigned short* __restrict__ Wk,
                      const unsigned short* __restrict__ Wv,
                      const unsigned short* __restrict__ Wr,
                      const float* __restrict__ sA, const float* __restrict__ sB,
                      const float* __restrict__ sp,
                      const float* __restrict__ tdec, const float* __restrict__ tfirst,
                      float* __restrict__ nA, float* __restrict__ nB,
                      float* __restrict__ np, unsigned short* __restrict__ rab) {
  extern __shared__ unsigned short dynlds[];   // 2 buffers x 6 tiles x 10KB = 120KB

  const int tid  = threadIdx.x;
  const int lane = tid & 31;
  const int wave = tid >> 5;
  const int wm = (wave & 3) * 32;   // wave tile: 32 (M) x 64 (N)
  const int wn = (wave >> 2) * 64;
  const int m0 = blockIdx.y * BM;
  const int n0 = blockIdx.x * BN;

  // Waves 0..5 each own one tile stream: {xk,xv,xr} rows m0.., {Wk,Wv,Wr} rows n0..
  const unsigned short* gsrc = nullptr;
  if (wave < 6) {
    const unsigned short* srcs[6] = { xk, xv, xr, Wk, Wv, Wr };
    int rbase = (wave < 3) ? m0 : n0;
    gsrc = srcs[wave] + (size_t)rbase * E_DIM;
  }

  v8f acc[3][2][4];
  const v8f vzero = { 0.f, 0.f, 0.f, 0.f, 0.f, 0.f, 0.f, 0.f };
#pragma unroll
  for (int g = 0; g < 3; ++g)
#pragma unroll
    for (int i = 0; i < 2; ++i)
#pragma unroll
      for (int j = 0; j < 4; ++j) acc[g][i][j] = vzero;

  // prologue: fill buffer 0
  if (wave < 6)
    tdm_load_tile(gsrc, (unsigned)(uintptr_t)(dynlds + wave * TILE_E));

  for (int kt = 0; kt < NK; ++kt) {
    const int cur = kt & 1;
    if (wave < 6) {
      if (kt + 1 < NK) { // prefetch next K-slab into the other buffer (TDM, async)
        tdm_load_tile(gsrc + (size_t)(kt + 1) * BK,
                      (unsigned)(uintptr_t)(dynlds + ((1 - cur) * 6 + wave) * TILE_E));
        __builtin_amdgcn_s_wait_tensorcnt(1);  // oldest (current buffer) done
      } else {
        __builtin_amdgcn_s_wait_tensorcnt(0);
      }
    }
    __syncthreads();     // current buffer visible to all waves

#pragma unroll
    for (int g = 0; g < 3; ++g) {
      const unsigned short* tA = dynlds + (cur * 6 + g)     * TILE_E;
      const unsigned short* tB = dynlds + (cur * 6 + 3 + g) * TILE_E;
      v16bf af[2], bf[4];
      af[0] = frag_a(tA, wm,      lane);
      af[1] = frag_a(tA, wm + 16, lane);
#pragma unroll
      for (int j = 0; j < 4; ++j) bf[j] = frag_b(tB, wn + 16 * j, lane);
#pragma unroll
      for (int i = 0; i < 2; ++i)
#pragma unroll
        for (int j = 0; j < 4; ++j)
          acc[g][i][j] = __builtin_amdgcn_wmma_f32_16x16x32_bf16(
              false, af[i], false, bf[j], (short)0, acc[g][i][j], false, false);
    }
    __syncthreads();     // compute done before TDM overwrites this buffer
  }

  // fused WKV epilogue. C/D layout: VGPR r -> M = r + (lane<16 ? 0 : 8), N = lane&15.
  const int nsel = lane & 15;
  const int msel = (lane >> 4) << 3;
#pragma unroll
  for (int j = 0; j < 4; ++j) {
    const int colE = n0 + wn + 16 * j + nsel;
    const float tf = tfirst[colE];
    const float td = -__expf(tdec[colE]);
#pragma unroll
    for (int i = 0; i < 2; ++i) {
      const int rowBase = m0 + wm + 16 * i + msel;
#pragma unroll
      for (int r = 0; r < 8; ++r) {
        const size_t idx = (size_t)(rowBase + r) * E_DIM + colE;
        const float kv = acc[0][i][j][r];
        const float vv = acc[1][i][j][r];
        const float rl = acc[2][i][j][r];
        const float rr = fminf(fmaxf(fmaf(rl, 0.16666667f, 0.5f), 0.0f), 1.0f);
        const float spv = sp[idx], sAv = sA[idx], sBv = sB[idx];
        // branch 1: output mix with time_first
        float ww = tf + kv;
        float p  = fmaxf(spv, ww);
        float e1 = __expf(spv - p);
        float e2 = __expf(ww - p);
        float a  = e1 * sAv + e2 * vv;
        float b  = e1 * sBv + e2;
        // branch 2: state decay
        float ww2 = spv + td;
        float p2  = fmaxf(ww2, kv);
        float e1b = __expf(ww2 - p2);
        float e2b = __expf(kv - p2);
        nA[idx] = e1b * sAv + e2b * vv;
        nB[idx] = e1b * sBv + e2b;
        np[idx] = p2;
        rab[idx] = f2bf(rr * a / b);
      }
    }
  }
}

// --------------------- output NT-GEMM: rwkv = rab @ Wo^T -------------------
__global__ __launch_bounds__(256, 1)
void gemm_out_kernel(const unsigned short* __restrict__ A,
                     const unsigned short* __restrict__ Bw,
                     float* __restrict__ C) {
  extern __shared__ unsigned short dynlds[];   // 2 buffers x 2 tiles x 10KB = 40KB

  const int tid  = threadIdx.x;
  const int lane = tid & 31;
  const int wave = tid >> 5;
  const int wm = (wave & 3) * 32;
  const int wn = (wave >> 2) * 64;
  const int m0 = blockIdx.y * BM;
  const int n0 = blockIdx.x * BN;

  const unsigned short* gsrc = nullptr;
  if (wave < 2)
    gsrc = (wave == 0) ? (A + (size_t)m0 * E_DIM) : (Bw + (size_t)n0 * E_DIM);

  v8f acc[2][4];
  const v8f vzero = { 0.f, 0.f, 0.f, 0.f, 0.f, 0.f, 0.f, 0.f };
#pragma unroll
  for (int i = 0; i < 2; ++i)
#pragma unroll
    for (int j = 0; j < 4; ++j) acc[i][j] = vzero;

  if (wave < 2)
    tdm_load_tile(gsrc, (unsigned)(uintptr_t)(dynlds + wave * TILE_E));

  for (int kt = 0; kt < NK; ++kt) {
    const int cur = kt & 1;
    if (wave < 2) {
      if (kt + 1 < NK) {
        tdm_load_tile(gsrc + (size_t)(kt + 1) * BK,
                      (unsigned)(uintptr_t)(dynlds + ((1 - cur) * 2 + wave) * TILE_E));
        __builtin_amdgcn_s_wait_tensorcnt(1);
      } else {
        __builtin_amdgcn_s_wait_tensorcnt(0);
      }
    }
    __syncthreads();

    {
      const unsigned short* tA = dynlds + (cur * 2 + 0) * TILE_E;
      const unsigned short* tB = dynlds + (cur * 2 + 1) * TILE_E;
      v16bf af[2], bf[4];
      af[0] = frag_a(tA, wm,      lane);
      af[1] = frag_a(tA, wm + 16, lane);
#pragma unroll
      for (int j = 0; j < 4; ++j) bf[j] = frag_b(tB, wn + 16 * j, lane);
#pragma unroll
      for (int i = 0; i < 2; ++i)
#pragma unroll
        for (int j = 0; j < 4; ++j)
          acc[i][j] = __builtin_amdgcn_wmma_f32_16x16x32_bf16(
              false, af[i], false, bf[j], (short)0, acc[i][j], false, false);
    }
    __syncthreads();
  }

  const int nsel = lane & 15;
  const int msel = (lane >> 4) << 3;
#pragma unroll
  for (int j = 0; j < 4; ++j) {
    const int colE = n0 + wn + 16 * j + nsel;
#pragma unroll
    for (int i = 0; i < 2; ++i) {
      const int rowBase = m0 + wm + 16 * i + msel;
#pragma unroll
      for (int r = 0; r < 8; ++r)
        C[(size_t)(rowBase + r) * E_DIM + colE] = acc[i][j][r];
    }
  }
}

// ------------------------------- launcher ----------------------------------
extern "C" void kernel_launch(void* const* d_in, const int* in_sizes, int n_in,
                              void* d_out, int out_size, void* d_ws, size_t ws_size,
                              hipStream_t stream) {
  const float* x      = (const float*)d_in[0];
  const float* sA     = (const float*)d_in[1];
  const float* sB     = (const float*)d_in[2];
  const float* sp     = (const float*)d_in[3];
  const float* sx     = (const float*)d_in[4];
  const float* tdec   = (const float*)d_in[5];
  const float* tfirst = (const float*)d_in[6];
  const float* tmk    = (const float*)d_in[7];
  const float* tmv    = (const float*)d_in[8];
  const float* tmr    = (const float*)d_in[9];
  const float* Wk     = (const float*)d_in[10];
  const float* Wv     = (const float*)d_in[11];
  const float* Wr     = (const float*)d_in[12];
  const float* Wo     = (const float*)d_in[13];

  const size_t BE = (size_t)B_DIM * E_DIM;
  const size_t EE = (size_t)E_DIM * E_DIM;

  float* rwkv = (float*)d_out;       // output order: rwkv, new_A, new_B, new_p, new_x
  float* nA   = rwkv + BE;
  float* nB   = nA + BE;
  float* np   = nB + BE;
  float* nx   = np + BE;

  unsigned short* w16 = (unsigned short*)d_ws;   // bf16 workspace
  unsigned short* xk  = w16;
  unsigned short* xv  = xk + BE;
  unsigned short* xr  = xv + BE;
  unsigned short* rab = xr + BE;
  unsigned short* WkB = rab + BE;
  unsigned short* WvB = WkB + EE;
  unsigned short* WrB = WvB + EE;
  unsigned short* WoB = WrB + EE;

  const size_t sh3 = (size_t)2 * 6 * TILE_E * sizeof(unsigned short);  // 122880 B
  const size_t sh1 = (size_t)2 * 2 * TILE_E * sizeof(unsigned short);  // 40960 B

  prep_kernel<<<(unsigned)(BE / 1024), 256, 0, stream>>>(x, sx, tmk, tmv, tmr, xk, xv, xr, nx);
  wconv_kernel<<<dim3((unsigned)(EE / 1024), 4), 256, 0, stream>>>(Wk, Wv, Wr, Wo, WkB, WvB, WrB, WoB);
  rwkv_fused_gemm3<<<dim3(E_DIM / BN, B_DIM / BM), 256, sh3, stream>>>(
      xk, xv, xr, WkB, WvB, WrB, sA, sB, sp, tdec, tfirst, nA, nB, np, rab);
  gemm_out_kernel<<<dim3(E_DIM / BN, B_DIM / BM), 256, sh1, stream>>>(rab, WoB, rwkv);
}
